// GCRN_37323265802601
// MI455X (gfx1250) — matrix operand order repaired and checked by reading, not compiled, not run on hardware
//
#include <hip/hip_runtime.h>

#define NND 1024
#define BB  64
#define TT  12
#define HH  64
#define CP  96
#define ROWS (NND * BB)          // 65536 (n-major, r = n*64 + b)
#define LDSS 40                   // LDS row stride (bf16 elems), 80B

typedef unsigned short ushort_t;
typedef __attribute__((ext_vector_type(16))) __bf16 v16bf;
typedef __attribute__((ext_vector_type(8)))  __bf16 v8bf;
typedef __attribute__((ext_vector_type(8)))  float  v8f;

__device__ __forceinline__ ushort_t f2bf(float x) {
  union { float f; unsigned u; } v; v.f = x;
  unsigned r = (v.u + 0x7FFFu + ((v.u >> 16) & 1u)) >> 16;
  return (ushort_t)r;
}

// CDNA5 async global->LDS copy (16B per lane), tracked by ASYNCcnt.
// LDS target address = low 32 bits of the generic shared-pointer (ISA 10.2:
// LDS aperture uses addr[31:0] directly).
__device__ __forceinline__ void async_copy16(unsigned lds_addr, const void* gaddr) {
  asm volatile("global_load_async_to_lds_b128 %0, %1, off"
               :: "v"(lds_addr), "v"(gaddr) : "memory");
}
__device__ __forceinline__ void wait_async0() {
  asm volatile("s_wait_asynccnt 0x0" ::: "memory");
}
__device__ __forceinline__ unsigned lds_addr_of(const void* p) {
  return (unsigned)(unsigned long long)p;
}

// ---------------------------------------------------------------------------
// bf16 WMMA GEMM: C[M,N] = A[M,K] * B[K,N], fp32 accumulate.
// PRECONDITION: M % 128 == 0, N % 128 == 0, K % 32 == 0 (no bounds checks).
// MODE 1: store bf16.  MODE 2: store bf16 of (2*acc - I)  (Chebyshev T2).
// 128x128 block tile, 8 waves (2x4), 64x32 per wave, double-buffered LDS:
// A tiles via async global->LDS, B tiles via register-staged transpose.
// ---------------------------------------------------------------------------
template<int MODE>
__global__ __launch_bounds__(256)
void gemm_bf16_kernel(const ushort_t* __restrict__ A, int lda,
                      const ushort_t* __restrict__ Bm, int ldb,
                      void* __restrict__ Cout, int ldc,
                      int Ktot)
{
  __shared__ ushort_t sA[2][128 * LDSS];
  __shared__ ushort_t sB[2][128 * LDSS];

  const int tid   = threadIdx.x;
  const int lane  = tid & 31;
  const int wave  = tid >> 5;
  const int wr    = wave >> 2;          // 0..1  (M blocks of 64)
  const int wc    = wave & 3;           // 0..3  (N blocks of 32)
  const int lhalf = (lane >> 4) & 1;
  const int l16   = lane & 15;

  const int m0 = blockIdx.y * 128;
  const int n0 = blockIdx.x * 128;

  // per-thread tile coordinates (A: 2 chunks of 8; B: 2 chunks of 8, same k)
  const int aRow0  = (tid * 2) >> 2;          // 0..127
  const int aCol0  = ((tid * 2) & 3) * 8;     // 0,8,16,24
  const int aRow1  = (tid * 2 + 1) >> 2;
  const int aCol1  = ((tid * 2 + 1) & 3) * 8;
  const int bKrow  = (tid * 2) >> 4;          // 0..31 (both chunks same k)
  const int bN0    = ((tid * 2) & 15) * 8;    // 0..112 (chunk1 = +8)

  const ushort_t* aP0 = A + (size_t)(m0 + aRow0) * lda + aCol0;
  const ushort_t* aP1 = A + (size_t)(m0 + aRow1) * lda + aCol1;
  const ushort_t* bP  = Bm + (size_t)bKrow * ldb + n0 + bN0;
  ushort_t* sA0 = &sA[0][aRow0 * LDSS + aCol0];
  ushort_t* sA1 = &sA[0][aRow1 * LDSS + aCol1];

  v8f acc[4][2];
  #pragma unroll
  for (int i = 0; i < 4; ++i)
    #pragma unroll
    for (int j = 0; j < 2; ++j)
      #pragma unroll
      for (int v = 0; v < 8; ++v) acc[i][j][v] = 0.0f;

  const int ktiles = Ktot >> 5;

  auto loadA = [&](int kt, int buf) {
    async_copy16(lds_addr_of(sA0 + buf * (128 * LDSS)), aP0 + kt * 32);
    async_copy16(lds_addr_of(sA1 + buf * (128 * LDSS)), aP1 + kt * 32);
  };
  auto loadB_regs = [&](int kt, ushort_t (&tmp)[2][8]) {
    const ushort_t* p = bP + (size_t)kt * 32 * ldb;
    *(uint4*)tmp[0] = *(const uint4*)p;
    *(uint4*)tmp[1] = *(const uint4*)(p + 8);
  };
  auto storeB = [&](int buf, ushort_t (&tmp)[2][8]) {
    #pragma unroll
    for (int c = 0; c < 2; ++c)
      #pragma unroll
      for (int e = 0; e < 8; ++e)
        sB[buf][(bN0 + c * 8 + e) * LDSS + bKrow] = tmp[c][e];
  };

  ushort_t btmp[2][8];
  loadA(0, 0);
  loadB_regs(0, btmp);
  storeB(0, btmp);
  wait_async0();
  __syncthreads();

  for (int kt = 0; kt < ktiles; ++kt) {
    const int cur = kt & 1, nxt = cur ^ 1;
    const bool more = (kt + 1 < ktiles);
    if (more) {
      loadA(kt + 1, nxt);                 // async, lands in alt buffer
      loadB_regs(kt + 1, btmp);           // global loads in flight over compute
    }

    v16bf af[4];
    #pragma unroll
    for (int mi = 0; mi < 4; ++mi) {
      int row = wr * 64 + mi * 16 + l16;
      int kb  = lhalf * 8;                // A frag: K {0-7,16-23}/{8-15,24-31}
      const __bf16* pa = (const __bf16*)&sA[cur][row * LDSS];
      v8bf lo = *(const v8bf*)(pa + kb);
      v8bf hi = *(const v8bf*)(pa + kb + 16);
      af[mi] = __builtin_shufflevector(lo, hi, 0,1,2,3,4,5,6,7,8,9,10,11,12,13,14,15);
    }
    v16bf bfr[2];
    #pragma unroll
    for (int ni = 0; ni < 2; ++ni) {
      int ncol = wc * 32 + ni * 16 + l16;
      int kb   = lhalf * 8;               // B frag: K {0-7,16-23}/{8-15,24-31}
      const __bf16* pb = (const __bf16*)&sB[cur][ncol * LDSS + kb];
      v8bf lo = *(const v8bf*)pb;
      v8bf hi = *(const v8bf*)(pb + 16);
      bfr[ni] = __builtin_shufflevector(lo, hi, 0,1,2,3,4,5,6,7,8,9,10,11,12,13,14,15);
    }
    #pragma unroll
    for (int mi = 0; mi < 4; ++mi)
      #pragma unroll
      for (int ni = 0; ni < 2; ++ni)
        acc[mi][ni] = __builtin_amdgcn_wmma_f32_16x16x32_bf16(
            false, af[mi], false, bfr[ni], (short)0, acc[mi][ni], false, false);

    if (more) {
      storeB(nxt, btmp);
      wait_async0();
    }
    __syncthreads();
  }

  #pragma unroll
  for (int mi = 0; mi < 4; ++mi)
    #pragma unroll
    for (int ni = 0; ni < 2; ++ni) {
      int gn = n0 + wc * 32 + ni * 16 + l16;
      #pragma unroll
      for (int v = 0; v < 8; ++v) {
        int gm = m0 + wr * 64 + mi * 16 + lhalf * 8 + v;
        float val = acc[mi][ni][v];
        if (MODE == 2) {
          float o = 2.0f * val - ((gm == gn) ? 1.0f : 0.0f);
          ((ushort_t*)Cout)[(size_t)gm * ldc + gn] = f2bf(o);
        } else {
          ((ushort_t*)Cout)[(size_t)gm * ldc + gn] = f2bf(val);
        }
      }
    }
}

// ---------------------------------------------------------------------------
// cat3 GEMM: C[ROWS,N] = [Xc|Z1|Z2] (each [ROWS,CP] bf16) @ Wpad[288,N] bf16
// fp32 output. K = 9 aligned tiles of 32 (fully unrolled -> constant
// addresses, no indexed-VGPR access). N in {64,128}, gridDim.x == 1.
// Out-of-range weight columns (N==64) zero-filled once in the prologue.
// ---------------------------------------------------------------------------
__global__ __launch_bounds__(256)
void gemm_cat3_kernel(const ushort_t* __restrict__ a0, const ushort_t* __restrict__ a1,
                      const ushort_t* __restrict__ a2, const ushort_t* __restrict__ Wm,
                      float* __restrict__ Cout, int N)
{
  __shared__ ushort_t sA[2][128 * LDSS];
  __shared__ ushort_t sB[2][128 * LDSS];

  const int tid   = threadIdx.x;
  const int lane  = tid & 31;
  const int wave  = tid >> 5;
  const int wr    = wave >> 2;
  const int wc    = wave & 3;
  const int lhalf = (lane >> 4) & 1;
  const int l16   = lane & 15;
  const int m0    = blockIdx.y * 128;

  const int aRow0 = (tid * 2) >> 2;
  const int aCol0 = ((tid * 2) & 3) * 8;
  const int aRow1 = (tid * 2 + 1) >> 2;
  const int aCol1 = ((tid * 2 + 1) & 3) * 8;
  const int bKrow = (tid * 2) >> 4;
  const int bN0   = ((tid * 2) & 15) * 8;

  const size_t aBase0 = (size_t)(m0 + aRow0) * CP + aCol0;
  const size_t aBase1 = (size_t)(m0 + aRow1) * CP + aCol1;
  ushort_t* sA0 = &sA[0][aRow0 * LDSS + aCol0];
  ushort_t* sA1 = &sA[0][aRow1 * LDSS + aCol1];
  const bool b0ok = (bN0 < N);            // chunk-level (N multiple of 8)
  const bool b1ok = (bN0 + 8 < N + 1) && (bN0 + 8 < 128) ? (bN0 + 8 < N) : false;

  // zero the permanently-out-of-range weight columns once (both buffers)
  #pragma unroll
  for (int c = 0; c < 2; ++c) {
    int gn = bN0 + c * 8;
    if (gn >= N) {
      #pragma unroll
      for (int e = 0; e < 8; ++e) {
        sB[0][(gn + e) * LDSS + bKrow] = 0;
        sB[1][(gn + e) * LDSS + bKrow] = 0;
      }
    }
  }

  v8f acc[4][2];
  #pragma unroll
  for (int i = 0; i < 4; ++i)
    #pragma unroll
    for (int j = 0; j < 2; ++j)
      #pragma unroll
      for (int v = 0; v < 8; ++v) acc[i][j][v] = 0.0f;

  auto loadA = [&](int kt, int buf) {
    const ushort_t* s = (kt < 3) ? a0 : ((kt < 6) ? a1 : a2);
    const int koff = (kt - ((kt < 3) ? 0 : ((kt < 6) ? 3 : 6))) * 32;
    async_copy16(lds_addr_of(sA0 + buf * (128 * LDSS)), s + aBase0 + koff);
    async_copy16(lds_addr_of(sA1 + buf * (128 * LDSS)), s + aBase1 + koff);
  };
  auto loadB_regs = [&](int kt, ushort_t (&tmp)[2][8]) {
    const int gk = kt * 32 + bKrow;       // always < 288
    #pragma unroll
    for (int c = 0; c < 2; ++c) {
      int gn = bN0 + c * 8;
      if (gn < N)
        *(uint4*)tmp[c] = *(const uint4*)(Wm + (size_t)gk * N + gn);
    }
  };
  auto storeB = [&](int buf, ushort_t (&tmp)[2][8]) {
    #pragma unroll
    for (int c = 0; c < 2; ++c) {
      int gn = bN0 + c * 8;
      if (gn < N) {
        #pragma unroll
        for (int e = 0; e < 8; ++e)
          sB[buf][(gn + e) * LDSS + bKrow] = tmp[c][e];
      }
    }
  };
  (void)b0ok; (void)b1ok;

  ushort_t btmp[2][8];
  loadA(0, 0);
  loadB_regs(0, btmp);
  storeB(0, btmp);
  wait_async0();
  __syncthreads();

  #pragma unroll
  for (int kt = 0; kt < 9; ++kt) {
    const int cur = kt & 1, nxt = cur ^ 1;
    const bool more = (kt + 1 < 9);
    if (more) {
      loadA(kt + 1, nxt);
      loadB_regs(kt + 1, btmp);
    }

    v16bf af[4];
    #pragma unroll
    for (int mi = 0; mi < 4; ++mi) {
      int row = wr * 64 + mi * 16 + l16;
      int kb  = lhalf * 8;
      const __bf16* pa = (const __bf16*)&sA[cur][row * LDSS];
      v8bf lo = *(const v8bf*)(pa + kb);
      v8bf hi = *(const v8bf*)(pa + kb + 16);
      af[mi] = __builtin_shufflevector(lo, hi, 0,1,2,3,4,5,6,7,8,9,10,11,12,13,14,15);
    }
    v16bf bfr[2];
    #pragma unroll
    for (int ni = 0; ni < 2; ++ni) {
      int ncol = wc * 32 + ni * 16 + l16;
      int kb   = lhalf * 8;
      const __bf16* pb = (const __bf16*)&sB[cur][ncol * LDSS + kb];
      v8bf lo = *(const v8bf*)pb;
      v8bf hi = *(const v8bf*)(pb + 16);
      bfr[ni] = __builtin_shufflevector(lo, hi, 0,1,2,3,4,5,6,7,8,9,10,11,12,13,14,15);
    }
    #pragma unroll
    for (int mi = 0; mi < 4; ++mi)
      #pragma unroll
      for (int ni = 0; ni < 2; ++ni)
        acc[mi][ni] = __builtin_amdgcn_wmma_f32_16x16x32_bf16(
            false, af[mi], false, bfr[ni], (short)0, acc[mi][ni], false, false);

    if (more) {
      storeB(nxt, btmp);
      wait_async0();
    }
    __syncthreads();
  }

  #pragma unroll
  for (int mi = 0; mi < 4; ++mi)
    #pragma unroll
    for (int ni = 0; ni < 2; ++ni) {
      int gn = wc * 32 + ni * 16 + l16;
      if (gn >= N) continue;
      #pragma unroll
      for (int v = 0; v < 8; ++v) {
        int gm = m0 + wr * 64 + mi * 16 + lhalf * 8 + v;
        Cout[(size_t)gm * N + gn] = acc[mi][ni][v];
      }
    }
}

// ---------------------------------------------------------------------------
// A = softmax(relu(E @ E^T)) per row, bf16 output. One block per row.
// ---------------------------------------------------------------------------
__global__ __launch_bounds__(256)
void adj_kernel(const float* __restrict__ emb, ushort_t* __restrict__ Abf)
{
  __shared__ float semb[NND * 8];
  __shared__ float red[256];
  const int tid = threadIdx.x;
  const int row = blockIdx.x;
  for (int i = tid; i < NND * 8; i += 256) semb[i] = emb[i];
  __syncthreads();
  float e[8];
  #pragma unroll
  for (int q = 0; q < 8; ++q) e[q] = semb[row * 8 + q];
  float sc[4], lmax = 0.0f;
  #pragma unroll
  for (int i = 0; i < 4; ++i) {
    int j = tid + i * 256;
    float s = 0.0f;
    #pragma unroll
    for (int q = 0; q < 8; ++q) s += e[q] * semb[j * 8 + q];
    s = fmaxf(s, 0.0f);
    sc[i] = s;
    lmax = fmaxf(lmax, s);
  }
  red[tid] = lmax; __syncthreads();
  for (int off = 128; off > 0; off >>= 1) {
    if (tid < off) red[tid] = fmaxf(red[tid], red[tid + off]);
    __syncthreads();
  }
  float mx = red[0]; __syncthreads();
  float lsum = 0.0f;
  #pragma unroll
  for (int i = 0; i < 4; ++i) { sc[i] = __expf(sc[i] - mx); lsum += sc[i]; }
  red[tid] = lsum; __syncthreads();
  for (int off = 128; off > 0; off >>= 1) {
    if (tid < off) red[tid] += red[tid + off];
    __syncthreads();
  }
  float inv = 1.0f / red[0];
  #pragma unroll
  for (int i = 0; i < 4; ++i)
    Abf[(size_t)row * NND + tid + i * 256] = f2bf(sc[i] * inv);
}

// ---- weights: [3*C, ncols] f32 -> zero-padded [3*CP, ncols] bf16 ----------
__global__ void pad_w_kernel(const float* __restrict__ W, ushort_t* __restrict__ Wp,
                             int C, int ncols)
{
  int idx = blockIdx.x * blockDim.x + threadIdx.x;
  int total = 3 * CP * ncols;
  if (idx >= total) return;
  int rp = idx / ncols, j = idx - rp * ncols;
  int k = rp / CP, c = rp - k * CP;
  float v = (c < C) ? W[(size_t)(k * C + c) * ncols + j] : 0.0f;
  Wp[idx] = f2bf(v);
}

__global__ void zero_kernel(float* __restrict__ p, int n)
{
  for (int i = blockIdx.x * blockDim.x + threadIdx.x; i < n; i += gridDim.x * blockDim.x)
    p[i] = 0.0f;
}

// ---- feature packers / GRU elementwise ------------------------------------
__global__ __launch_bounds__(256)
void pack_enc_kernel(const float* __restrict__ x, const float* __restrict__ h,
                     ushort_t* __restrict__ Xc, int t)
{
  int r = blockIdx.x * 256 + threadIdx.x;
  int n = r >> 6, b = r & 63;
  ushort_t* row = Xc + (size_t)r * CP;
  row[0] = f2bf(x[((size_t)b * TT + t) * NND + n]);
  const float* hr = h + (size_t)r * HH;
  #pragma unroll
  for (int j = 0; j < HH; ++j) row[1 + j] = f2bf(hr[j]);
  #pragma unroll
  for (int j = 1 + HH; j < CP; ++j) row[j] = 0;
}

__global__ __launch_bounds__(256)
void gate_enc_kernel(const float* __restrict__ G, const float* __restrict__ gb,
                     const float* __restrict__ x, const float* __restrict__ h,
                     float* __restrict__ R, ushort_t* __restrict__ Xc, int t)
{
  int r = blockIdx.x * 256 + threadIdx.x;
  int n = r >> 6, b = r & 63;
  const float* gr = G + (size_t)r * 128;
  const float* hr = h + (size_t)r * HH;
  float* Rr = R + (size_t)r * HH;
  ushort_t* row = Xc + (size_t)r * CP;
  row[0] = f2bf(x[((size_t)b * TT + t) * NND + n]);
  #pragma unroll
  for (int j = 0; j < HH; ++j) {
    float z  = 1.0f / (1.0f + __expf(-(gr[j] + gb[j])));
    float rr = 1.0f / (1.0f + __expf(-(gr[HH + j] + gb[HH + j])));
    Rr[j] = rr;
    row[1 + j] = f2bf(z * hr[j]);
  }
  #pragma unroll
  for (int j = 1 + HH; j < CP; ++j) row[j] = 0;
}

__global__ __launch_bounds__(256)
void pack_dec_kernel(const float* __restrict__ GO, const float* __restrict__ yc,
                     const float* __restrict__ h, ushort_t* __restrict__ Xc, int t)
{
  int r = blockIdx.x * 256 + threadIdx.x;
  int n = r >> 6, b = r & 63;
  ushort_t* row = Xc + (size_t)r * CP;
  row[0] = f2bf(GO[r]);
  row[1] = f2bf(yc[((size_t)b * TT + t) * NND + n]);
  const float* hr = h + (size_t)r * HH;
  #pragma unroll
  for (int j = 0; j < HH; ++j) row[2 + j] = f2bf(hr[j]);
  #pragma unroll
  for (int j = 2 + HH; j < CP; ++j) row[j] = 0;
}

__global__ __launch_bounds__(256)
void gate_dec_kernel(const float* __restrict__ G, const float* __restrict__ gb,
                     const float* __restrict__ GO, const float* __restrict__ yc,
                     const float* __restrict__ h, float* __restrict__ R,
                     ushort_t* __restrict__ Xc, int t)
{
  int r = blockIdx.x * 256 + threadIdx.x;
  int n = r >> 6, b = r & 63;
  const float* gr = G + (size_t)r * 128;
  const float* hr = h + (size_t)r * HH;
  float* Rr = R + (size_t)r * HH;
  ushort_t* row = Xc + (size_t)r * CP;
  row[0] = f2bf(GO[r]);
  row[1] = f2bf(yc[((size_t)b * TT + t) * NND + n]);
  #pragma unroll
  for (int j = 0; j < HH; ++j) {
    float z  = 1.0f / (1.0f + __expf(-(gr[j] + gb[j])));
    float rr = 1.0f / (1.0f + __expf(-(gr[HH + j] + gb[HH + j])));
    Rr[j] = rr;
    row[2 + j] = f2bf(z * hr[j]);
  }
  #pragma unroll
  for (int j = 2 + HH; j < CP; ++j) row[j] = 0;
}

__global__ __launch_bounds__(256)
void update_kernel(const float* __restrict__ G, const float* __restrict__ ub,
                   const float* __restrict__ R, float* __restrict__ h,
                   float* __restrict__ GO, float* __restrict__ out,
                   const float* __restrict__ pW, const float* __restrict__ pb,
                   int t, int doProj)
{
  int r = blockIdx.x * 256 + threadIdx.x;
  int n = r >> 6, b = r & 63;
  const float* gr = G + (size_t)r * HH;
  const float* Rr = R + (size_t)r * HH;
  float* hr = h + (size_t)r * HH;
  float accp = 0.0f;
  #pragma unroll
  for (int j = 0; j < HH; ++j) {
    float hc = tanhf(gr[j] + ub[j]);
    float rr = Rr[j];
    float hn = rr * hr[j] + (1.0f - rr) * hc;
    hr[j] = hn;
    accp += hn * pW[j];
  }
  if (doProj) {
    float go = accp + pb[0];
    GO[r] = go;
    out[((size_t)b * TT + t) * NND + n] = go;
  }
}

// ---------------------------------------------------------------------------
extern "C" void kernel_launch(void* const* d_in, const int* in_sizes, int n_in,
                              void* d_out, int out_size, void* d_ws, size_t ws_size,
                              hipStream_t stream)
{
  (void)in_sizes; (void)n_in; (void)out_size; (void)ws_size;
  const float* x    = (const float*)d_in[0];
  const float* ycov = (const float*)d_in[1];
  const float* emb  = (const float*)d_in[2];
  const float* egW  = (const float*)d_in[3];
  const float* egb  = (const float*)d_in[4];
  const float* euW  = (const float*)d_in[5];
  const float* eub  = (const float*)d_in[6];
  const float* dgW  = (const float*)d_in[7];
  const float* dgb  = (const float*)d_in[8];
  const float* duW  = (const float*)d_in[9];
  const float* dub  = (const float*)d_in[10];
  const float* pW   = (const float*)d_in[11];
  const float* pb   = (const float*)d_in[12];
  float* out = (float*)d_out;

  char* base = (char*)d_ws;
  size_t off = 0;
  auto alloc = [&](size_t bytes) -> char* {
    char* p = base + off;
    off += (bytes + 255) & ~(size_t)255;
    return p;
  };
  ushort_t* Abf  = (ushort_t*)alloc((size_t)NND * NND * 2);
  ushort_t* S2   = (ushort_t*)alloc((size_t)NND * NND * 2);
  ushort_t* Xc   = (ushort_t*)alloc((size_t)ROWS * CP * 2);
  ushort_t* Z1   = (ushort_t*)alloc((size_t)ROWS * CP * 2);
  ushort_t* Z2   = (ushort_t*)alloc((size_t)ROWS * CP * 2);
  float*    G    = (float*)alloc((size_t)ROWS * 128 * 4);
  float*    Rg   = (float*)alloc((size_t)ROWS * HH * 4);
  float*    Hst  = (float*)alloc((size_t)ROWS * HH * 4);
  float*    GO   = (float*)alloc((size_t)ROWS * 4);
  ushort_t* Wpeg = (ushort_t*)alloc(3 * CP * 128 * 2);
  ushort_t* Wpeu = (ushort_t*)alloc(3 * CP * 64 * 2);
  ushort_t* Wpdg = (ushort_t*)alloc(3 * CP * 128 * 2);
  ushort_t* Wpdu = (ushort_t*)alloc(3 * CP * 64 * 2);

  const int MCOL = BB * CP;  // 6144

  // ---- setup: adjacency, T2 support, padded weights, zeroed state --------
  adj_kernel<<<NND, 256, 0, stream>>>(emb, Abf);
  {
    dim3 g(NND / 128, NND / 128);
    gemm_bf16_kernel<2><<<g, 256, 0, stream>>>(Abf, NND, Abf, NND, (void*)S2, NND, NND);
  }
  pad_w_kernel<<<(3 * CP * 128 + 255) / 256, 256, 0, stream>>>(egW, Wpeg, 65, 128);
  pad_w_kernel<<<(3 * CP * 64 + 255) / 256, 256, 0, stream>>>(euW, Wpeu, 65, 64);
  pad_w_kernel<<<(3 * CP * 128 + 255) / 256, 256, 0, stream>>>(dgW, Wpdg, 66, 128);
  pad_w_kernel<<<(3 * CP * 64 + 255) / 256, 256, 0, stream>>>(duW, Wpdu, 66, 64);
  zero_kernel<<<1024, 256, 0, stream>>>(Hst, ROWS * HH);
  zero_kernel<<<64, 256, 0, stream>>>(GO, ROWS);

  dim3 g1(MCOL / 128, NND / 128);   // graph GEMM: 1024 x 6144 x 1024
  dim3 g2(1, ROWS / 128);           // weight GEMM: 65536 x {128,64} x 288
  const int eb = ROWS / 256;

  for (int t = 0; t < TT; ++t) {    // encoder
    pack_enc_kernel<<<eb, 256, 0, stream>>>(x, Hst, Xc, t);
    gemm_bf16_kernel<1><<<g1, 256, 0, stream>>>(Abf, NND, Xc, MCOL, (void*)Z1, MCOL, NND);
    gemm_bf16_kernel<1><<<g1, 256, 0, stream>>>(S2,  NND, Xc, MCOL, (void*)Z2, MCOL, NND);
    gemm_cat3_kernel<<<g2, 256, 0, stream>>>(Xc, Z1, Z2, Wpeg, G, 128);
    gate_enc_kernel<<<eb, 256, 0, stream>>>(G, egb, x, Hst, Rg, Xc, t);
    gemm_bf16_kernel<1><<<g1, 256, 0, stream>>>(Abf, NND, Xc, MCOL, (void*)Z1, MCOL, NND);
    gemm_bf16_kernel<1><<<g1, 256, 0, stream>>>(S2,  NND, Xc, MCOL, (void*)Z2, MCOL, NND);
    gemm_cat3_kernel<<<g2, 256, 0, stream>>>(Xc, Z1, Z2, Wpeu, G, 64);
    update_kernel<<<eb, 256, 0, stream>>>(G, eub, Rg, Hst, GO, out, pW, pb, t, 0);
  }
  for (int t = 0; t < TT; ++t) {    // decoder
    pack_dec_kernel<<<eb, 256, 0, stream>>>(GO, ycov, Hst, Xc, t);
    gemm_bf16_kernel<1><<<g1, 256, 0, stream>>>(Abf, NND, Xc, MCOL, (void*)Z1, MCOL, NND);
    gemm_bf16_kernel<1><<<g1, 256, 0, stream>>>(S2,  NND, Xc, MCOL, (void*)Z2, MCOL, NND);
    gemm_cat3_kernel<<<g2, 256, 0, stream>>>(Xc, Z1, Z2, Wpdg, G, 128);
    gate_dec_kernel<<<eb, 256, 0, stream>>>(G, dgb, GO, ycov, Hst, Rg, Xc, t);
    gemm_bf16_kernel<1><<<g1, 256, 0, stream>>>(Abf, NND, Xc, MCOL, (void*)Z1, MCOL, NND);
    gemm_bf16_kernel<1><<<g1, 256, 0, stream>>>(S2,  NND, Xc, MCOL, (void*)Z2, MCOL, NND);
    gemm_cat3_kernel<<<g2, 256, 0, stream>>>(Xc, Z1, Z2, Wpdu, G, 64);
    update_kernel<<<eb, 256, 0, stream>>>(G, dub, Rg, Hst, GO, out, pW, pb, t, 1);
  }
}
